// NER_CRF_LSTM_29497835389502
// MI455X (gfx1250) — compile-verified, matrix operand order checked by reading, not run on hardware
//
#include <hip/hip_runtime.h>
#include <hip/hip_bf16.h>

// ---------------------------------------------------------------------------
// BiLSTM-CRF NER pipeline for gfx1250 (MI455X).
//   K1 embed gather -> bf16
//   K2 W_ih cast -> bf16
//   K3 input-projection GEMM [8192,256]x[256,2048] via v_wmma_f32_16x16x32_bf16
//   K4 serial LSTM recurrence (1 WGP per direction, W_hh streamed from L2)
//   K5 output projection (f32 VALU)
//   K6 Viterbi forward + backtrack (one wave, backpointers in 160KB LDS)
// ---------------------------------------------------------------------------

typedef __attribute__((ext_vector_type(16))) __bf16 v16bf;
typedef __attribute__((ext_vector_type(8)))  __bf16 v8bf;
typedef __attribute__((ext_vector_type(8)))  float  v8f;

#define SEQ   8192
#define EMBD  256
#define HDIM  256     // per-direction hidden
#define G4    1024    // 4*HDIM gate rows per direction
#define COLS  2048    // both directions' gates
#define HID2  512     // concat hidden
#define NTAGS 20
#define STARTT 18
#define STOPT  19
#define NEGV (-10000.0f)

// ---------------- K1: embedding gather + cast to bf16 ----------------------
__global__ void embed_cast_kernel(const int* __restrict__ x,
                                  const float* __restrict__ emb,
                                  __bf16* __restrict__ xe) {
  const int t = blockIdx.x;
  const int j = threadIdx.x;
  const int id = x[t];
  xe[t * EMBD + j] = (__bf16)emb[id * EMBD + j];
}

// ---------------- K2: W_ih (both dirs) cast to bf16 ------------------------
__global__ void wcast_kernel(const float* __restrict__ wf,
                             const float* __restrict__ wb,
                             __bf16* __restrict__ o) {
  const int r = blockIdx.x;   // 0..2047
  const int j = threadIdx.x;  // 0..255
  const float v = (r < G4) ? wf[r * EMBD + j] : wb[(r - G4) * EMBD + j];
  o[r * EMBD + j] = (__bf16)v;
}

// ---------------- K3: pre = xe @ W_ih^T + b  (WMMA bf16, f32 accum) --------
// block = (32,8): 8 waves, each wave owns one 16x16 tile.
// grid  = (COLS/16/8, SEQ/16) = (16, 512)
__global__ void __launch_bounds__(256)
gemm_pre_kernel(const __bf16* __restrict__ A,   // [SEQ, EMBD]   row-major
                const __bf16* __restrict__ B,   // [COLS, EMBD]  row-major (= W_ih)
                const float* __restrict__ bias_f,
                const float* __restrict__ bias_b,
                float* __restrict__ C) {        // [SEQ, COLS]
  const int lane = threadIdx.x;          // 0..31
  const int wave = threadIdx.y;          // 0..7
  const int tm   = blockIdx.y;           // M tile
  const int tn   = blockIdx.x * 8 + wave;// N tile
  const int half = lane >> 4;            // 0 | 1
  const int l15  = lane & 15;
  const int kbase = half * 8;            // lanes 16..31 take K chunks +8 / +24

  const __bf16* arow = A + (tm * 16 + l15) * EMBD;  // row M = l15
  const __bf16* brow = B + (tn * 16 + l15) * EMBD;  // col N = l15 (B = W^T)

  v8f acc = {};
  for (int kb = 0; kb < EMBD; kb += 32) {
    v8bf a0 = *(const v8bf*)(arow + kb + kbase);        // K = kb+kbase .. +7
    v8bf a1 = *(const v8bf*)(arow + kb + kbase + 16);   // K = kb+kbase+16 ..
    v8bf b0 = *(const v8bf*)(brow + kb + kbase);
    v8bf b1 = *(const v8bf*)(brow + kb + kbase + 16);
    v16bf av, bv;
#pragma unroll
    for (int i = 0; i < 8; ++i) {
      av[i] = a0[i]; av[8 + i] = a1[i];
      bv[i] = b0[i]; bv[8 + i] = b1[i];
    }
    acc = __builtin_amdgcn_wmma_f32_16x16x32_bf16(
        /*neg_a=*/false, av, /*neg_b=*/false, bv,
        /*c_mod=*/(short)0, acc, /*reuse_a=*/false, /*reuse_b=*/false);
  }

  const int coln = tn * 16 + l15;
  const float bias = (coln < G4) ? bias_f[coln] : bias_b[coln - G4];
  float* crow = C + (tm * 16 + half * 8) * COLS + coln;  // M = tm*16 + half*8 + r
#pragma unroll
  for (int r = 0; r < 8; ++r) crow[r * COLS] = acc[r] + bias;
}

// ---------------- K4: serial LSTM, one block per direction -----------------
__device__ __forceinline__ float sigf(float x) { return 1.0f / (1.0f + __expf(-x)); }

__global__ void __launch_bounds__(1024)
lstm_kernel(const float* __restrict__ pre,     // [SEQ, COLS]
            const float* __restrict__ whh_f,   // [G4, HDIM]
            const float* __restrict__ whh_b,   // [G4, HDIM]
            float* __restrict__ hs) {          // [SEQ, HID2]  (fwd | bwd)
  const int dir = blockIdx.x;                  // 0 = fwd, 1 = bwd
  const int tid = threadIdx.x;                 // 0..1023 -> gate row
  const float* wrow = (dir ? whh_b : whh_f) + tid * HDIM;  // L2-resident row

  __shared__ float hbuf[HDIM];
  __shared__ float gbuf[G4];
  if (tid < HDIM) hbuf[tid] = 0.0f;
  float cst = 0.0f;
  __syncthreads();

  for (int s = 0; s < SEQ; ++s) {
    const int t = dir ? (SEQ - 1 - s) : s;
    float acc = pre[t * COLS + dir * G4 + tid];
#pragma unroll 4
    for (int k = 0; k < HDIM; k += 4) {
      const float4 w = *(const float4*)(wrow + k);
      const float4 h = *(const float4*)(hbuf + k);
      acc += w.x * h.x + w.y * h.y + w.z * h.z + w.w * h.w;
    }
    gbuf[tid] = acc;
    __syncthreads();
    if (tid < HDIM) {
      const float ig = sigf(gbuf[tid]);
      const float fg = sigf(gbuf[HDIM + tid]);
      const float gg = tanhf(gbuf[2 * HDIM + tid]);
      const float og = sigf(gbuf[3 * HDIM + tid]);
      cst = fg * cst + ig * gg;
      const float h = og * tanhf(cst);
      hbuf[tid] = h;
      hs[t * HID2 + dir * HDIM + tid] = h;
    }
    __syncthreads();
  }
}

// ---------------- K5: feats = hs @ w_out^T + b_out -------------------------
__global__ void __launch_bounds__(64)
proj_kernel(const float* __restrict__ hs, const float* __restrict__ wo,
            const float* __restrict__ bo, float* __restrict__ feats) {
  __shared__ float sh[HID2];
  const int t = blockIdx.x;
  const int tid = threadIdx.x;  // 64
  for (int k = tid; k < HID2; k += 64) sh[k] = hs[t * HID2 + k];
  __syncthreads();
  if (tid < NTAGS) {
    float acc = bo[tid];
    const float* wr = wo + tid * HID2;
    for (int k = 0; k < HID2; k += 4) {
      const float4 w = *(const float4*)(wr + k);
      acc += w.x * sh[k] + w.y * sh[k + 1] + w.z * sh[k + 2] + w.w * sh[k + 3];
    }
    feats[t * NTAGS + tid] = acc;
  }
}

// ---------------- K6: Viterbi forward + backtrack (one wave) ---------------
// Backpointers kept in LDS (8192*20 bytes = 160 KB <= 320 KB/WGP) so the
// serial backtrack chain runs at LDS latency.
__global__ void __launch_bounds__(32)
viterbi_kernel(const float* __restrict__ feats, const float* __restrict__ trans,
               float* __restrict__ out) {
  __shared__ unsigned char bp[SEQ * NTAGS];
  __shared__ float strans[NTAGS * NTAGS];
  __shared__ float fwd[NTAGS];
  const int tid = threadIdx.x;

  for (int i = tid; i < NTAGS * NTAGS; i += 32) strans[i] = trans[i];
  if (tid < NTAGS) fwd[tid] = (tid == STARTT) ? 0.0f : NEGV;
  __syncthreads();

  for (int t = 0; t < SEQ; ++t) {
    const float ft = (tid < NTAGS) ? feats[t * NTAGS + tid] : 0.0f;
    float best = -3.0e38f;
    int bi = 0;
    if (tid < NTAGS) {
      const float* tr = strans + tid * NTAGS;  // transition[next=tid, prev]
      for (int p = 0; p < NTAGS; ++p) {
        const float sc = fwd[p] + tr[p];
        if (sc > best) { best = sc; bi = p; }  // strict '>' == first-max argmax
      }
    }
    __syncthreads();
    if (tid < NTAGS) {
      fwd[tid] = best + ft;
      bp[t * NTAGS + tid] = (unsigned char)bi;
    }
    __syncthreads();
  }

  if (tid == 0) {
    float best = -3.0e38f;
    int bi = 0;
    for (int p = 0; p < NTAGS; ++p) {
      const float sc = fwd[p] + strans[STOPT * NTAGS + p];
      if (sc > best) { best = sc; bi = p; }
    }
    out[0] = best;                     // score (f32)
    int* path = (int*)(out + 1);       // path as int32 payload
    int tag = bi;
    for (int t = SEQ - 1; t >= 0; --t) {
      path[t] = tag;
      tag = (int)bp[t * NTAGS + tag];
    }
  }
}

// ---------------------------------------------------------------------------
extern "C" void kernel_launch(void* const* d_in, const int* in_sizes, int n_in,
                              void* d_out, int out_size, void* d_ws, size_t ws_size,
                              hipStream_t stream) {
  const int*   x      = (const int*)d_in[0];
  const float* emb    = (const float*)d_in[1];
  const float* w_ih_f = (const float*)d_in[2];
  const float* w_hh_f = (const float*)d_in[3];
  const float* b_f    = (const float*)d_in[4];
  const float* w_ih_b = (const float*)d_in[5];
  const float* w_hh_b = (const float*)d_in[6];
  const float* b_b    = (const float*)d_in[7];
  const float* w_out  = (const float*)d_in[8];
  const float* b_out  = (const float*)d_in[9];
  const float* trans  = (const float*)d_in[10];
  float* out = (float*)d_out;

  // workspace layout
  char* ws = (char*)d_ws;
  __bf16* xe   = (__bf16*)(ws + 0);                       //  4 MB  [SEQ, EMBD]
  __bf16* wbf  = (__bf16*)(ws + (size_t)4 * 1024 * 1024); //  1 MB  [COLS, EMBD]
  float*  pre  = (float*) (ws + (size_t)5 * 1024 * 1024); // 64 MB  [SEQ, COLS]
  float*  hs   = (float*) (ws + (size_t)69 * 1024 * 1024);// 16 MB  [SEQ, HID2]
  float*  feats= (float*) (ws + (size_t)85 * 1024 * 1024);// .7 MB  [SEQ, NTAGS]

  embed_cast_kernel<<<SEQ, EMBD, 0, stream>>>(x, emb, xe);
  wcast_kernel<<<COLS, EMBD, 0, stream>>>(w_ih_f, w_ih_b, wbf);
  gemm_pre_kernel<<<dim3(COLS / 16 / 8, SEQ / 16), dim3(32, 8), 0, stream>>>(
      xe, wbf, b_f, b_b, pre);
  lstm_kernel<<<2, 1024, 0, stream>>>(pre, w_hh_f, w_hh_b, hs);
  proj_kernel<<<SEQ, 64, 0, stream>>>(hs, w_out, b_out, feats);
  viterbi_kernel<<<1, 32, 0, stream>>>(feats, trans, out);
}
